// LSTMChar_7481833030292
// MI455X (gfx1250) — matrix operand order repaired and checked by reading, not compiled, not run on hardware
//
#include <hip/hip_runtime.h>

typedef __attribute__((ext_vector_type(16))) _Float16 v16h;
typedef __attribute__((ext_vector_type(8)))  _Float16 v8h;
typedef __attribute__((ext_vector_type(8)))  float    v8f;

constexpr int H        = 128;         // hidden
constexpr int IN       = 69;          // input features
constexpr int KPAD     = 224;         // 128 (h) + 96 (x padded from 69)
constexpr int NS       = KPAD / 32;   // 7 WMMA k-steps
constexpr int MB       = 16;          // batch rows per workgroup (WMMA M tile)
constexpr int NTHREADS = 512;         // 16 waves: 4 gates x (2 hidden tiles each)

__device__ __forceinline__ float fast_sig(float x) {
    return __builtin_amdgcn_rcpf(1.0f + __expf(-x));
}
__device__ __forceinline__ float fast_tanh(float x) {
    float e = __expf(2.0f * x);
    return 1.0f - 2.0f * __builtin_amdgcn_rcpf(e + 1.0f);
}

__global__ __launch_bounds__(NTHREADS, 1)
void lstm_fused_wmma(const float* __restrict__ x,
                     const float* __restrict__ W_ih,
                     const float* __restrict__ W_hh,
                     const float* __restrict__ b_ih,
                     const float* __restrict__ b_hh,
                     const float* __restrict__ W_out,
                     const float* __restrict__ b_out,
                     float* __restrict__ out,
                     int T)
{
    // A staging: rows = 16 batch rows, cols 0..127 = h_{t-1}, 128..196 = x_t, 197..223 = 0
    __shared__ alignas(16) _Float16 sh[MB * KPAD];          // 7 KB
    // activated f,g,o published by waves q=1..3 for the c/h update
    __shared__ float gbuf[3][8][32][8];                     // 24 KB

    const int tid  = threadIdx.x;
    const int wave = tid >> 5;          // 0..15
    const int lane = tid & 31;
    const int q    = wave >> 2;         // gate: 0=i 1=f 2=g 3=o
    const int wp   = wave & 3;          // tile pair selector
    const int w0   = wp;                // hidden tiles 0..3
    const int w1   = wp + 4;            // hidden tiles 4..7
    const int half = lane >> 4;         // K-half selector per WMMA layouts
    const int ln16 = lane & 15;
    const int mblk = blockIdx.x * MB;   // batch row base

    // ---- preload this wave's B fragments: columns of [W_hh | W_ih | 0] in f16 ----
    const int gcol0 = q * H + w0 * 16 + ln16;  // gate rows (i,f,g,o order)
    const int gcol1 = q * H + w1 * 16 + ln16;
    v16h bfrag0[NS], bfrag1[NS];
#pragma unroll
    for (int s = 0; s < NS; ++s) {
        const int kbase = s * 32 + half * 8;  // B lane holds K chunks [k0+8*half,+8) and [k0+16+8*half,+8) of one column
#pragma unroll
        for (int j = 0; j < 16; ++j) {
            const int k = kbase + j + (j >= 8 ? 8 : 0);
            float v0 = (k < H) ? W_hh[gcol0 * H + k]
                               : ((k - H) < IN ? W_ih[gcol0 * IN + (k - H)] : 0.0f);
            float v1 = (k < H) ? W_hh[gcol1 * H + k]
                               : ((k - H) < IN ? W_ih[gcol1 * IN + (k - H)] : 0.0f);
            bfrag0[s][j] = (_Float16)v0;
            bfrag1[s][j] = (_Float16)v1;
        }
    }
    const float bias0 = b_ih[gcol0] + b_hh[gcol0];  // folded into accumulator init
    const float bias1 = b_ih[gcol1] + b_hh[gcol1];

    // zero the A tile (h starts at 0; K-padding stays 0 forever)
    for (int i = tid; i < MB * KPAD; i += NTHREADS) sh[i] = (_Float16)0.0f;
    __syncthreads();
    // stage x_0
    for (int idx = tid; idx < MB * IN; idx += NTHREADS) {
        const int m = idx / IN, i = idx - m * IN;
        sh[m * KPAD + H + i] = (_Float16)x[((size_t)(mblk + m) * T + 0) * IN + i];
    }
    __syncthreads();

    v8f cst0 = {0.f,0.f,0.f,0.f,0.f,0.f,0.f,0.f};  // cell state (q==0 waves)
    v8f cst1 = {0.f,0.f,0.f,0.f,0.f,0.f,0.f,0.f};

    for (int t = 0; t < T; ++t) {
        // ---- phase 1: gates = [h|x_t] @ [W_hh|W_ih]^T + b ----
        // preload ALL A fragments: 14 ds_load_b128 back-to-back, staged waits
        v16h afrag[NS];
#pragma unroll
        for (int s = 0; s < NS; ++s) {
            // 16-bit A 16x32 layout: row = lane%16; K chunks [k0+8*half,+8) and [k0+16+8*half,+8)
            const _Float16* rp = &sh[ln16 * KPAD + s * 32 + half * 8];
            v8h alo = *(const v8h*)rp;
            v8h ahi = *(const v8h*)(rp + 16);
            afrag[s] = __builtin_shufflevector(alo, ahi,
                        0,1,2,3,4,5,6,7,8,9,10,11,12,13,14,15);
        }
        // keep the load group ahead of the WMMA chain
        __builtin_amdgcn_sched_barrier(0);

        // two independent accumulation chains: one per hidden tile (WMMA ILP)
        v8f acc0 = {bias0,bias0,bias0,bias0,bias0,bias0,bias0,bias0};
        v8f acc1 = {bias1,bias1,bias1,bias1,bias1,bias1,bias1,bias1};
#pragma unroll
        for (int s = 0; s < NS; ++s) {
            acc0 = __builtin_amdgcn_wmma_f32_16x16x32_f16(
                       false, afrag[s], false, bfrag0[s], (short)0, acc0, false, false);
            acc1 = __builtin_amdgcn_wmma_f32_16x16x32_f16(
                       false, afrag[s], false, bfrag1[s], (short)0, acc1, false, false);
        }

        float act0[8], act1[8];
        if (q == 2) {
#pragma unroll
            for (int j = 0; j < 8; ++j) { act0[j] = fast_tanh(acc0[j]);
                                          act1[j] = fast_tanh(acc1[j]); }
        } else {
#pragma unroll
            for (int j = 0; j < 8; ++j) { act0[j] = fast_sig(acc0[j]);
                                          act1[j] = fast_sig(acc1[j]); }
        }
        if (q != 0) {
#pragma unroll
            for (int j = 0; j < 8; ++j) {
                gbuf[q - 1][w0][lane][j] = act0[j];
                gbuf[q - 1][w1][lane][j] = act1[j];
            }
        }
        __syncthreads();   // all A reads + gate publishes done

        // ---- phase 2: c/h update (q==0 waves) overlapped with x_{t+1} staging ----
        if (q == 0) {
#pragma unroll
            for (int j = 0; j < 8; ++j) {
                // tile w0
                {
                    const float fg = gbuf[0][w0][lane][j];
                    const float gg = gbuf[1][w0][lane][j];
                    const float og = gbuf[2][w0][lane][j];
                    const float c  = fg * cst0[j] + act0[j] * gg;  // act0 = sigmoid(i)
                    cst0[j] = c;
                    const float hv = og * fast_tanh(c);
                    // C/D layout: M = j + 8*half, N = lane%16 -> h column w*16 + ln16
                    sh[(j + half * 8) * KPAD + w0 * 16 + ln16] = (_Float16)hv;
                }
                // tile w1
                {
                    const float fg = gbuf[0][w1][lane][j];
                    const float gg = gbuf[1][w1][lane][j];
                    const float og = gbuf[2][w1][lane][j];
                    const float c  = fg * cst1[j] + act1[j] * gg;
                    cst1[j] = c;
                    const float hv = og * fast_tanh(c);
                    sh[(j + half * 8) * KPAD + w1 * 16 + ln16] = (_Float16)hv;
                }
            }
        } else if (t + 1 < T) {
            for (int idx = tid - 128; idx < MB * IN; idx += (NTHREADS - 128)) {
                const int m = idx / IN, i = idx - m * IN;
                sh[m * KPAD + H + i] =
                    (_Float16)x[((size_t)(mblk + m) * T + (t + 1)) * IN + i];
            }
        }
        __syncthreads();   // new h / x visible before next step's reads
    }

    // ---- head: out[b][cls] = h_T[b] . W_out[cls] + b_out[cls] ----
    if (tid < MB * 2) {
        const int m = tid >> 1, cls = tid & 1;
        float s = b_out[cls];
#pragma unroll
        for (int k = 0; k < H; ++k)
            s += (float)sh[m * KPAD + k] * W_out[cls * H + k];
        out[(mblk + m) * 2 + cls] = s;
    }
}

extern "C" void kernel_launch(void* const* d_in, const int* in_sizes, int n_in,
                              void* d_out, int out_size, void* d_ws, size_t ws_size,
                              hipStream_t stream) {
    const float* x     = (const float*)d_in[0];
    const float* W_ih  = (const float*)d_in[1];
    const float* W_hh  = (const float*)d_in[2];
    const float* b_ih  = (const float*)d_in[3];
    const float* b_hh  = (const float*)d_in[4];
    const float* W_out = (const float*)d_in[5];
    const float* b_out = (const float*)d_in[6];
    float* out = (float*)d_out;

    const int B = out_size / 2;                 // 256
    const int T = in_sizes[0] / (B * IN);       // 1024

    dim3 grid(B / MB), block(NTHREADS);
    lstm_fused_wmma<<<grid, block, 0, stream>>>(x, W_ih, W_hh, b_ih, b_hh,
                                                W_out, b_out, out, T);
}
